// ThorFNN_30520037606034
// MI455X (gfx1250) — matrix operand order repaired and checked by reading, not compiled, not run on hardware
//
#include <hip/hip_runtime.h>

// ThorFNN eval path on MI455X (gfx1250): grouped MoE GEMM with bf16 WMMA.
// out[token] = relu(x[token] @ W1[e] + b1[e]) @ W2[e] + b2[e], e = mask[token]
// Double-buffered LDS (1 barrier / K-tile), packed b32 DS stores for the
// fp32->bf16 weight tiles, wave32 WMMA 16x16x32 bf16 with f32 accumulation.

#define E_ 8
#define S_ 2048
#define B_ 4
#define D_ 1024
#define F_ 4096
#define NTOK (S_ * B_)     // 8192 tokens
#define KBLK 64
#define LDSTR 72           // padded LDS row stride (elements)

typedef __bf16 bf16_t;
typedef __attribute__((ext_vector_type(16))) bf16_t v16bf;
typedef __attribute__((ext_vector_type(8)))  float  v8f;

union FragU { uint4 q[2]; v16bf v; };

__device__ __forceinline__ unsigned short f32_bf16(float f) {
    unsigned int u = __float_as_uint(f);
    unsigned int r = (u + 0x7FFFu + ((u >> 16) & 1u)) >> 16;  // round-to-nearest-even
    return (unsigned short)r;
}

__device__ __forceinline__ uint4 pack8_bf16(float4 a, float4 b) {
    uint4 r;
    r.x = (unsigned)f32_bf16(a.x) | ((unsigned)f32_bf16(a.y) << 16);
    r.y = (unsigned)f32_bf16(a.z) | ((unsigned)f32_bf16(a.w) << 16);
    r.z = (unsigned)f32_bf16(b.x) | ((unsigned)f32_bf16(b.y) << 16);
    r.w = (unsigned)f32_bf16(b.z) | ((unsigned)f32_bf16(b.w) << 16);
    return r;
}

// ---------------- token routing ----------------

__global__ void k_zero(int* counts, int* fill) {
    int t = threadIdx.x;
    if (t < E_) { counts[t] = 0; fill[t] = 0; }
}

__global__ void k_count(const int* __restrict__ mask, int* counts) {
    int i = blockIdx.x * blockDim.x + threadIdx.x;
    if (i < NTOK) atomicAdd(&counts[mask[i]], 1);
}

__global__ void k_scan(const int* __restrict__ counts, int* offsets) {
    if (threadIdx.x == 0) {
        int acc = 0;
        for (int e = 0; e < E_; ++e) { offsets[e] = acc; acc += counts[e]; }
        offsets[E_] = acc;
    }
}

__global__ void k_scatter(const int* __restrict__ mask, const int* __restrict__ offsets,
                          int* fill, int* __restrict__ perm) {
    int i = blockIdx.x * blockDim.x + threadIdx.x;
    if (i < NTOK) {
        int e = mask[i];
        int pos = offsets[e] + atomicAdd(&fill[e], 1);
        perm[pos] = i;   // expert-sorted token list
    }
}

// gather x rows in sorted order, fp32 -> packed bf16 (8 elems / thread)
__global__ void k_gather(const float* __restrict__ x, const int* __restrict__ perm,
                         uint4* __restrict__ xb) {
    int t = blockIdx.x * blockDim.x + threadIdx.x;   // NTOK*D/8 threads
    int row = t >> 7;                                // D/8 = 128 chunks per row
    int c8  = t & 127;
    int src = perm[row];
    const float4* p = (const float4*)(x + (size_t)src * D_ + c8 * 8);
    xb[(size_t)row * (D_ / 8) + c8] = pack8_bf16(p[0], p[1]);
}

// ---------------- shared WMMA GEMM core (double-buffered LDS) ----------------
// A: bf16 [rowsValid][ASTR] (pre-offset to rowBase), W: fp32 [KDIM][BSTR].
// Block tile 128x128, KBLK=64 (2 WMMA k-steps / tile), 8 waves in 4x2 layout,
// each wave computes a 32x64 sub-tile = 2x4 WMMA accumulators.

template<int KDIM, int ASTR, int BSTR>
__device__ __forceinline__ void moe_gemm_core(
    const unsigned short* __restrict__ A,
    const float* __restrict__ W,
    int rowsValid, int colBase,
    unsigned short Al[2][128 * LDSTR],
    unsigned short Bl[2][128 * LDSTR],
    v8f acc[2][4])
{
    const int tid   = threadIdx.x;
    const int lane  = tid & 31;
    const int wave  = tid >> 5;
    const int waveM = (wave >> 1) * 32;   // 4 waves down (rows)
    const int waveN = (wave & 1) * 64;    // 2 waves across (cols)
    const int lm    = lane & 15;
    const int kh    = lane >> 4;

    const int ar  = tid >> 1;             // A-load: row 0..127
    const int ah  = (tid & 1) * 32;       //          32-element half row
    const int kp2 = (tid >> 3) * 2;       // B-load: k pair base 0,2,..,62
    const int bf0 = (tid & 7) * 16;       //          16-column chunk

    uint4 va[4];
    float wb0[16], wb1[16];

    auto gload = [&](int k0) {
        if (ar < rowsValid) {
            const uint4* s = (const uint4*)(A + (size_t)ar * ASTR + k0 + ah);
            va[0] = s[0]; va[1] = s[1]; va[2] = s[2]; va[3] = s[3];
        } else {
            va[0] = va[1] = va[2] = va[3] = make_uint4(0, 0, 0, 0);
        }
        const float4* s0 = (const float4*)(W + (size_t)(k0 + kp2) * BSTR + colBase + bf0);
        const float4* s1 = (const float4*)(W + (size_t)(k0 + kp2 + 1) * BSTR + colBase + bf0);
        #pragma unroll
        for (int j = 0; j < 4; ++j) {
            ((float4*)wb0)[j] = s0[j];
            ((float4*)wb1)[j] = s1[j];
        }
    };
    auto sstore = [&](int p) {
        uint4* d = (uint4*)(Al[p] + ar * LDSTR + ah);
        d[0] = va[0]; d[1] = va[1]; d[2] = va[2]; d[3] = va[3];
        #pragma unroll
        for (int j = 0; j < 16; ++j) {    // bf16 pair along K -> one b32 DS store
            unsigned u = (unsigned)f32_bf16(wb0[j]) | ((unsigned)f32_bf16(wb1[j]) << 16);
            *(unsigned*)(Bl[p] + (bf0 + j) * LDSTR + kp2) = u;
        }
    };

    gload(0);
    sstore(0);
    int p = 0;
    for (int k0 = 0; k0 < KDIM; k0 += KBLK) {
        __syncthreads();                  // buf[p] ready; buf[p^1] reads drained
        const bool nxt = (k0 + KBLK) < KDIM;
        if (nxt) gload(k0 + KBLK);        // prefetch next tile into registers

        #pragma unroll
        for (int kk = 0; kk < KBLK; kk += 32) {
            FragU af[2], bfg[4];
            #pragma unroll
            for (int m = 0; m < 2; ++m) {  // A: lanes 0-15 K=0..15, 16-31 K=16..31
                const unsigned short* q = Al[p] + (waveM + m * 16 + lm) * LDSTR + kk + kh * 16;
                af[m].q[0] = *(const uint4*)q;
                af[m].q[1] = *(const uint4*)(q + 8);
            }
            #pragma unroll
            for (int n = 0; n < 4; ++n) {  // B: lanes 0-15 K=0..15, 16-31 K=16..31
                const unsigned short* q = Bl[p] + (waveN + n * 16 + lm) * LDSTR + kk + kh * 16;
                bfg[n].q[0] = *(const uint4*)q;
                bfg[n].q[1] = *(const uint4*)(q + 8);
            }
            #pragma unroll
            for (int m = 0; m < 2; ++m)
                #pragma unroll
                for (int n = 0; n < 4; ++n)
                    acc[m][n] = __builtin_amdgcn_wmma_f32_16x16x32_bf16(
                        false, af[m].v, false, bfg[n].v, (short)0, acc[m][n], false, false);
        }

        if (nxt) sstore(p ^ 1);           // drain prefetch into the other buffer
        p ^= 1;
    }
}

// ---------------- GEMM1: h = relu(xs @ W1[e] + b1[e]), bf16 out ----------------

__global__ __launch_bounds__(256) void k_gemm1(
    const unsigned short* __restrict__ xb, const float* __restrict__ W1,
    const float* __restrict__ b1, const int* __restrict__ offsets,
    unsigned short* __restrict__ h)
{
    __shared__ __align__(16) unsigned short Al[2][128 * LDSTR];
    __shared__ __align__(16) unsigned short Bl[2][128 * LDSTR];

    const int e        = blockIdx.z;
    const int segStart = offsets[e];
    const int segEnd   = offsets[e + 1];
    const int rowBase  = segStart + blockIdx.y * 128;
    if (rowBase >= segEnd) return;
    const int rowsValid = segEnd - rowBase;
    const int fBase = blockIdx.x * 128;

    v8f acc[2][4];
    #pragma unroll
    for (int m = 0; m < 2; ++m)
        #pragma unroll
        for (int n = 0; n < 4; ++n)
            acc[m][n] = (v8f){0.f, 0.f, 0.f, 0.f, 0.f, 0.f, 0.f, 0.f};

    moe_gemm_core<D_, D_, F_>(xb + (size_t)rowBase * D_,
                              W1 + (size_t)e * D_ * F_,
                              rowsValid, fBase, Al, Bl, acc);

    // epilogue: + b1, relu, bf16 store. C layout: VGPR r -> M=r+8*kh, N=lm
    const int lane  = threadIdx.x & 31;
    const int wave  = threadIdx.x >> 5;
    const int waveM = (wave >> 1) * 32;
    const int waveN = (wave & 1) * 64;
    const int lm    = lane & 15;
    const int kh    = lane >> 4;

    #pragma unroll
    for (int n = 0; n < 4; ++n) {
        const int col  = fBase + waveN + n * 16 + lm;
        const float bi = b1[e * F_ + col];
        #pragma unroll
        for (int m = 0; m < 2; ++m) {
            #pragma unroll
            for (int r = 0; r < 8; ++r) {
                int row = waveM + m * 16 + r + kh * 8;
                if (row < rowsValid) {
                    float v = fmaxf(acc[m][n][r] + bi, 0.0f);
                    h[(size_t)(rowBase + row) * F_ + col] = f32_bf16(v);
                }
            }
        }
    }
}

// ---------------- GEMM2: out = h @ W2[e] + b2[e], scatter fp32 ----------------

__global__ __launch_bounds__(256) void k_gemm2(
    const unsigned short* __restrict__ h, const float* __restrict__ W2,
    const float* __restrict__ b2, const int* __restrict__ offsets,
    const int* __restrict__ perm, float* __restrict__ out)
{
    __shared__ __align__(16) unsigned short Al[2][128 * LDSTR];
    __shared__ __align__(16) unsigned short Bl[2][128 * LDSTR];

    const int e        = blockIdx.z;
    const int segStart = offsets[e];
    const int segEnd   = offsets[e + 1];
    const int rowBase  = segStart + blockIdx.y * 128;
    if (rowBase >= segEnd) return;
    const int rowsValid = segEnd - rowBase;
    const int dBase = blockIdx.x * 128;

    v8f acc[2][4];
    #pragma unroll
    for (int m = 0; m < 2; ++m)
        #pragma unroll
        for (int n = 0; n < 4; ++n)
            acc[m][n] = (v8f){0.f, 0.f, 0.f, 0.f, 0.f, 0.f, 0.f, 0.f};

    moe_gemm_core<F_, F_, D_>(h + (size_t)rowBase * F_,
                              W2 + (size_t)e * F_ * D_,
                              rowsValid, dBase, Al, Bl, acc);

    const int lane  = threadIdx.x & 31;
    const int wave  = threadIdx.x >> 5;
    const int waveM = (wave >> 1) * 32;
    const int waveN = (wave & 1) * 64;
    const int lm    = lane & 15;
    const int kh    = lane >> 4;

    #pragma unroll
    for (int n = 0; n < 4; ++n) {
        const int col  = dBase + waveN + n * 16 + lm;
        const float bi = b2[e * D_ + col];
        #pragma unroll
        for (int m = 0; m < 2; ++m) {
            #pragma unroll
            for (int r = 0; r < 8; ++r) {
                int row = waveM + m * 16 + r + kh * 8;
                if (row < rowsValid) {
                    int tok = perm[rowBase + row];
                    out[(size_t)tok * D_ + col] = acc[m][n][r] + bi;
                }
            }
        }
    }
}

// ---------------- launcher ----------------

extern "C" void kernel_launch(void* const* d_in, const int* in_sizes, int n_in,
                              void* d_out, int out_size, void* d_ws, size_t ws_size,
                              hipStream_t stream) {
    const float* x    = (const float*)d_in[0];
    const float* W1   = (const float*)d_in[1];
    const float* b1   = (const float*)d_in[2];
    const float* W2   = (const float*)d_in[3];
    const float* b2   = (const float*)d_in[4];
    const int*   mask = (const int*)d_in[5];
    float* out = (float*)d_out;

    // workspace layout (~80.1 MB)
    char* ws = (char*)d_ws;
    int* counts  = (int*)(ws + 0);
    int* fill    = (int*)(ws + 64);
    int* offsets = (int*)(ws + 128);
    int* perm    = (int*)(ws + 256);
    constexpr size_t permEnd = 256 + (size_t)NTOK * 4;                 // 33024
    unsigned short* xb = (unsigned short*)(ws + permEnd);              // NTOK*D bf16
    constexpr size_t xbEnd = permEnd + (size_t)NTOK * D_ * 2;          // +16 MB
    unsigned short* hbuf = (unsigned short*)(ws + xbEnd);              // NTOK*F bf16 (+64 MB)

    k_zero   <<<1, 32, 0, stream>>>(counts, fill);
    k_count  <<<NTOK / 256, 256, 0, stream>>>(mask, counts);
    k_scan   <<<1, 32, 0, stream>>>(counts, offsets);
    k_scatter<<<NTOK / 256, 256, 0, stream>>>(mask, offsets, fill, perm);
    k_gather <<<(NTOK * (D_ / 8)) / 256, 256, 0, stream>>>(x, perm, (uint4*)xb);
    k_gemm1  <<<dim3(F_ / 128, 64, E_), 256, 0, stream>>>(xb, W1, b1, offsets, hbuf);
    k_gemm2  <<<dim3(D_ / 128, 64, E_), 256, 0, stream>>>(hbuf, W2, b2, offsets, perm, out);
}
